// SMARTAgentDecoder_44676249813398
// MI455X (gfx1250) — matrix operand order repaired and checked by reading, not compiled, not run on hardware
//
#include <hip/hip_runtime.h>
#include <hip/hip_bf16.h>
#include <math.h>

typedef __attribute__((ext_vector_type(16))) _Float16 v16h;
typedef __attribute__((ext_vector_type(8)))  _Float16 v8h;
typedef __attribute__((ext_vector_type(2)))  _Float16 v2h;
typedef __attribute__((ext_vector_type(8)))  float    v8f;

#define PI_F     3.14159265358979323846f
#define TWO_PI_F 6.28318530717958647692f

enum { FLAG_ACC = 1, FLAG_RELU = 2, FLAG_F16OUT = 4 };

// ---------------------------------------------------------------------------
// WMMA GEMM: C[N,NC] = op(A[N,lda] @ B[K,NC] + bias).
// Requirements (guaranteed by callers): N%64==0, NC%32==0, K%32==0,
// lda a multiple of 8 elements (16B-aligned fragment loads).
// Each wave computes a 16x32 tile (2 accumulators); B k-slab staged in LDS
// as f16 shared by the 4 waves of the block. Guard-free inner loop.
// ---------------------------------------------------------------------------
__device__ __forceinline__ void load_a_frag(const float* __restrict__ p, int akb, v16h& af) {
  const float4 x0 = *(const float4*)(p + akb);
  const float4 x1 = *(const float4*)(p + akb + 4);
  const float4 x2 = *(const float4*)(p + akb + 16);
  const float4 x3 = *(const float4*)(p + akb + 20);
  af[0]=(_Float16)x0.x; af[1]=(_Float16)x0.y; af[2] =(_Float16)x0.z; af[3] =(_Float16)x0.w;
  af[4]=(_Float16)x1.x; af[5]=(_Float16)x1.y; af[6] =(_Float16)x1.z; af[7] =(_Float16)x1.w;
  af[8]=(_Float16)x2.x; af[9]=(_Float16)x2.y; af[10]=(_Float16)x2.z; af[11]=(_Float16)x2.w;
  af[12]=(_Float16)x3.x; af[13]=(_Float16)x3.y; af[14]=(_Float16)x3.z; af[15]=(_Float16)x3.w;
}
__device__ __forceinline__ void load_a_frag(const _Float16* __restrict__ p, int akb, v16h& af) {
  v8h h0 = *(const v8h*)(p + akb);
  v8h h1 = *(const v8h*)(p + akb + 16);
#pragma unroll
  for (int j = 0; j < 8; ++j) { af[j] = h0[j]; af[8 + j] = h1[j]; }
}

template <typename AT>
__global__ __launch_bounds__(128)
void gemm_wmma_kernel(const AT* __restrict__ A, int lda,
                      const float* __restrict__ B,
                      const float* __restrict__ bias,
                      float* __restrict__ C,
                      int N, int K, int NC, int flags)
{
  __shared__ _Float16 Bs[32 * 32];          // 2 KB k-slab: 32 K-rows x 32 cols
  const int tid  = threadIdx.x;
  const int lane = tid & 31;
  const int wave = tid >> 5;
  const int row0 = (blockIdx.x * 4 + wave) * 16;
  const int col0 = blockIdx.y * 32;
  const int m    = lane & 15;
  // 16-bit A 16x32 layout: lanes 0-15 -> K {0..7,16..23}; lanes 16-31 -> {8..15,24..31}
  const int akb  = (lane & 16) ? 8 : 0;
  // 16-bit B 32x16 layout: lanes 0-15 -> K {0..7,16..23}; lanes 16-31 -> {8..15,24..31}; col = lane&15
  const int bkb  = (lane & 16) ? 8 : 0;
  const AT* Ar = A + (size_t)(row0 + m) * lda;
  v8f acc0 = {}, acc1 = {};
  for (int k0 = 0; k0 < K; k0 += 32) {
    __syncthreads();                        // previous slab fully consumed
#pragma unroll
    for (int i = 0; i < 4; ++i) {           // 512 dwords (1024 f16) / 128 threads
      int p2 = tid + i * 128;               // dword index into Bs
      int kk = p2 >> 4;                     // 16 dwords per 32-col row
      int cc = (p2 & 15) * 2;
      float2 bb = *(const float2*)(B + (size_t)(k0 + kk) * NC + col0 + cc);
      v2h hp; hp[0] = (_Float16)bb.x; hp[1] = (_Float16)bb.y;
      ((v2h*)Bs)[p2] = hp;
    }
    __syncthreads();
    v16h af;
    load_a_frag(Ar + k0, akb, af);
    v16h bf0, bf1;
#pragma unroll
    for (int j = 0; j < 16; ++j) {
      bf0[j] = Bs[(bkb + j + (j >= 8 ? 8 : 0)) * 32 + m];
      bf1[j] = Bs[(bkb + j + (j >= 8 ? 8 : 0)) * 32 + 16 + m];
    }
    acc0 = __builtin_amdgcn_wmma_f32_16x16x32_f16(false, af, false, bf0, (short)0, acc0, false, false);
    acc1 = __builtin_amdgcn_wmma_f32_16x16x32_f16(false, af, false, bf1, (short)0, acc1, false, false);
  }
  // C/D layout: VGPR r -> row row0+r (lanes 0-15) / row0+8+r (lanes 16-31)
  const int rbase = row0 + ((lane & 16) ? 8 : 0);
  const int ca = col0 + m, cb = col0 + 16 + m;
  const float bva = bias ? bias[ca] : 0.0f;
  const float bvb = bias ? bias[cb] : 0.0f;
#pragma unroll
  for (int r = 0; r < 8; ++r) {
    size_t o0 = (size_t)(rbase + r) * NC + ca;
    size_t o1 = (size_t)(rbase + r) * NC + cb;
    float va = acc0[r] + bva, vb = acc1[r] + bvb;
    if (flags & FLAG_RELU) { va = fmaxf(va, 0.0f); vb = fmaxf(vb, 0.0f); }
    if (flags & FLAG_F16OUT) {
      ((_Float16*)C)[o0] = (_Float16)va; ((_Float16*)C)[o1] = (_Float16)vb;
    } else if (flags & FLAG_ACC) {
      C[o0] += va; C[o1] += vb;
    } else {
      C[o0] = va; C[o1] = vb;
    }
  }
}

// ---------------------------------------------------------------------------
// LayerNorm over rows of width 128 (optionally ReLU). 8 rows / 256-thread block.
// ---------------------------------------------------------------------------
template <typename TIN, typename TOUT>
__global__ __launch_bounds__(256)
void ln128_kernel(const TIN* __restrict__ x, const float* __restrict__ g,
                  const float* __restrict__ b, TOUT* __restrict__ y,
                  int N, int relu)
{
  const int lane = threadIdx.x & 31;
  const int row  = blockIdx.x * 8 + (threadIdx.x >> 5);
  if (row >= N) return;
  const TIN* xr = x + (size_t)row * 128;
  float v0 = (float)xr[lane],      v1 = (float)xr[lane + 32];
  float v2 = (float)xr[lane + 64], v3 = (float)xr[lane + 96];
  float s = v0 + v1 + v2 + v3;
  for (int msk = 1; msk < 32; msk <<= 1) s += __shfl_xor(s, msk);
  float mu = s * (1.0f / 128.0f);
  float d0 = v0 - mu, d1 = v1 - mu, d2 = v2 - mu, d3 = v3 - mu;
  float q = d0*d0 + d1*d1 + d2*d2 + d3*d3;
  for (int msk = 1; msk < 32; msk <<= 1) q += __shfl_xor(q, msk);
  float rstd = rsqrtf(q * (1.0f / 128.0f) + 1e-5f);
  TOUT* yr = y + (size_t)row * 128;
  float o0 = d0 * rstd * g[lane]      + b[lane];
  float o1 = d1 * rstd * g[lane + 32] + b[lane + 32];
  float o2 = d2 * rstd * g[lane + 64] + b[lane + 64];
  float o3 = d3 * rstd * g[lane + 96] + b[lane + 96];
  if (relu) { o0 = fmaxf(o0,0.f); o1 = fmaxf(o1,0.f); o2 = fmaxf(o2,0.f); o3 = fmaxf(o3,0.f); }
  yr[lane] = (TOUT)o0; yr[lane+32] = (TOUT)o1; yr[lane+64] = (TOUT)o2; yr[lane+96] = (TOUT)o3;
}

// ---------------------------------------------------------------------------
// Graph attention core: one wave per destination node; dst[] is sorted
// (true for all three edge sets in this workload). Deterministic (no atomics).
// cnt is wave-uniform so the i<cnt guards are scalar branches, not exec masks.
// ---------------------------------------------------------------------------
#define EDGE_CAP 16
__global__ __launch_bounds__(256)
void attn_gather_kernel(const float* __restrict__ q, const float* __restrict__ k,
                        const float* __restrict__ v, const float* __restrict__ kr,
                        const float* __restrict__ vr,
                        const int* __restrict__ src, const int* __restrict__ dst,
                        float* __restrict__ agg, int E, int Nd)
{
  const int lane = threadIdx.x & 31;
  const int node = blockIdx.x * 8 + (threadIdx.x >> 5);
  if (node >= Nd) return;
  int lo = 0, hi = E;
  while (lo < hi) { int mid = (lo + hi) >> 1; if (dst[mid] < node)     lo = mid + 1; else hi = mid; }
  int lo2 = lo, hi2 = E;
  while (lo2 < hi2){ int mid = (lo2 + hi2) >> 1; if (dst[mid] < node+1) lo2 = mid + 1; else hi2 = mid; }
  int cnt = lo2 - lo; if (cnt > EDGE_CAP) cnt = EDGE_CAP;
  const int dbase = lane * 4;                       // 4 dims/lane; head = lane>>2
  float4 qv = *(const float4*)(q + (size_t)node * 128 + dbase);
  float sims[EDGE_CAP];
  float mx = -3.0e38f;
#pragma unroll
  for (int i = 0; i < EDGE_CAP; ++i) {
    if (i < cnt) {
      int e = lo + i; int sN = src[e];
      float4 kv = *(const float4*)(k  + (size_t)sN * 128 + dbase);
      float4 rv = *(const float4*)(kr + (size_t)e  * 128 + dbase);
      float p = qv.x*(kv.x+rv.x) + qv.y*(kv.y+rv.y) + qv.z*(kv.z+rv.z) + qv.w*(kv.w+rv.w);
      p += __shfl_xor(p, 1); p += __shfl_xor(p, 2);  // 16-dim head dot complete
      p *= 0.25f;                                    // Dh^-0.5 = 1/4
      sims[i] = p; mx = fmaxf(mx, p);
    }
  }
  float den = 0.0f;
#pragma unroll
  for (int i = 0; i < EDGE_CAP; ++i)
    if (i < cnt) { float e2 = expf(sims[i] - mx); sims[i] = e2; den += e2; }
  float inv = 1.0f / (den + 1e-12f);
  float ax = 0.f, ay = 0.f, az = 0.f, aw = 0.f;
#pragma unroll
  for (int i = 0; i < EDGE_CAP; ++i) {
    if (i < cnt) {
      int e = lo + i; int sN = src[e];
      float w = sims[i] * inv;
      float4 vv = *(const float4*)(v  + (size_t)sN * 128 + dbase);
      float4 rv = *(const float4*)(vr + (size_t)e  * 128 + dbase);
      ax += w*(vv.x+rv.x); ay += w*(vv.y+rv.y); az += w*(vv.z+rv.z); aw += w*(vv.w+rv.w);
    }
  }
  float4 o; o.x = ax; o.y = ay; o.z = az; o.w = aw;
  *(float4*)(agg + (size_t)node * 128 + dbase) = o;
}

// ------------------------- elementwise / feature kernels -------------------
__device__ __forceinline__ float wrap_angle(float a) {
  float x = fmodf(a + PI_F, TWO_PI_F);
  if (x < 0.f) x += TWO_PI_F;
  return x - PI_F;
}
__device__ __forceinline__ float ang2(float cx, float cy, float nx, float ny) {
  float cr = cx * ny - cy * nx, dt = cx * nx + cy * ny;
  bool z = (cr == 0.f) && (dt == 0.f);
  return atan2f(z ? 0.f : cr, z ? 1.f : dt);
}
__device__ __forceinline__ float snorm2(float x, float y) {
  float s = x * x + y * y; return (s == 0.f) ? 0.f : sqrtf(s);
}

__global__ void feat_xa_kernel(const float* __restrict__ pos, const float* __restrict__ head,
                               float* __restrict__ feats, int NA, int T_) {
  int n = blockIdx.x * blockDim.x + threadIdx.x; if (n >= NA) return;
  int t = n % T_;
  float mx = 0.f, my = 0.f;
  if (t > 0) { mx = pos[2*n] - pos[2*(n-1)]; my = pos[2*n+1] - pos[2*(n-1)+1]; }
  float h = head[n], cx = cosf(h), cy = sinf(h);
  feats[2*n]   = snorm2(mx, my);
  feats[2*n+1] = ang2(cx, cy, mx, my);
}
__global__ void feat_rt_kernel(const float* __restrict__ pos, const float* __restrict__ head,
                               const int* __restrict__ src, const int* __restrict__ dst,
                               float* __restrict__ feats, int E, int T_) {
  int e = blockIdx.x * blockDim.x + threadIdx.x; if (e >= E) return;
  int s = src[e], d = dst[e];
  float rx = pos[2*s] - pos[2*d], ry = pos[2*s+1] - pos[2*d+1];
  float hd = head[d], cx = cosf(hd), cy = sinf(hd);
  feats[4*e+0] = snorm2(rx, ry);
  feats[4*e+1] = ang2(cx, cy, rx, ry);
  feats[4*e+2] = wrap_angle(head[s] - hd);
  feats[4*e+3] = (float)(d % T_) - (float)(s % T_);
}
__global__ void feat_pt_kernel(const float* __restrict__ mpos, const float* __restrict__ morient,
                               const float* __restrict__ pos, const float* __restrict__ head,
                               const int* __restrict__ src, const int* __restrict__ dst,
                               float* __restrict__ feats, int E, int A_, int T_) {
  int e = blockIdx.x * blockDim.x + threadIdx.x; if (e >= E) return;
  int s = src[e], d = dst[e];
  int nam = (d % A_) * T_ + (d / A_);                 // time-major -> agent-major
  float rx = mpos[2*s] - pos[2*nam], ry = mpos[2*s+1] - pos[2*nam+1];
  float hd = head[nam], cx = cosf(hd), cy = sinf(hd);
  feats[3*e+0] = snorm2(rx, ry);
  feats[3*e+1] = ang2(cx, cy, rx, ry);
  feats[3*e+2] = wrap_angle(morient[s] - hd);
}
__global__ void feat_aa_kernel(const float* __restrict__ pos, const float* __restrict__ head,
                               const int* __restrict__ src, const int* __restrict__ dst,
                               float* __restrict__ feats, int E, int A_, int T_) {
  int e = blockIdx.x * blockDim.x + threadIdx.x; if (e >= E) return;
  int s = src[e], d = dst[e];
  int sn = (s % A_) * T_ + (s / A_);
  int dn = (d % A_) * T_ + (d / A_);
  float rx = pos[2*sn] - pos[2*dn], ry = pos[2*sn+1] - pos[2*dn+1];
  float hd = head[dn], cx = cosf(hd), cy = sinf(hd);
  feats[3*e+0] = snorm2(rx, ry);
  feats[3*e+1] = ang2(cx, cy, rx, ry);
  feats[3*e+2] = wrap_angle(head[sn] - hd);
}
// z = [cos(x f 2pi) (64) | sin (64) | x | zero-pad to 160] per channel, f16
__global__ void fourier_z_kernel(const float* __restrict__ feats, const float* __restrict__ freqs,
                                 _Float16* __restrict__ z, int N, int C, int ch, int total) {
  int i = blockIdx.x * blockDim.x + threadIdx.x; if (i >= total) return;
  int n = i / 160, k = i % 160;
  float x = feats[(size_t)n * C + ch];
  float val;
  if      (k < 64)   val = cosf(x * freqs[ch * 64 + k] * TWO_PI_F);
  else if (k < 128)  val = sinf(x * freqs[ch * 64 + (k - 64)] * TWO_PI_F);
  else if (k == 128) val = x;
  else               val = 0.0f;
  z[i] = (_Float16)val;
}
// zero-pad trailing rows of a row-major [Ksrc,NC] weight into [Kdst,NC]
__global__ void pad_rows_kernel(const float* __restrict__ src, float* __restrict__ dst,
                                int total_src, int total_dst) {
  int i = blockIdx.x * blockDim.x + threadIdx.x; if (i >= total_dst) return;
  dst[i] = (i < total_src) ? src[i] : 0.0f;
}
// zero-pad trailing cols: [rows,scols] -> [rows,dcols]
__global__ void pad_cols_kernel(const float* __restrict__ src, float* __restrict__ dst,
                                int scols, int dcols, int total) {
  int i = blockIdx.x * blockDim.x + threadIdx.x; if (i >= total) return;
  int r = i / dcols, c = i % dcols;
  dst[i] = (c < scols) ? src[(size_t)r * scols + c] : 0.0f;
}
__global__ void add_cat_kernel(float* __restrict__ h, const float* __restrict__ cat,
                               int total, int T_) {
  int i = blockIdx.x * blockDim.x + threadIdx.x; if (i >= total) return;
  int n = i >> 7;
  h[i] += cat[((size_t)(n / T_) << 7) + (i & 127)];
}
__global__ void tok_gather_kernel(const float* __restrict__ tv, const float* __restrict__ tp,
                                  const float* __restrict__ tc, const int* __restrict__ tok_idx,
                                  const int* __restrict__ atype, float* __restrict__ out,
                                  int total, int T_) {
  int i = blockIdx.x * blockDim.x + threadIdx.x; if (i >= total) return;
  int n = i >> 7, a = n / T_;
  int ty = atype[a];
  const float* tab = (ty == 0) ? tv : ((ty == 1) ? tp : tc);
  out[i] = tab[((size_t)tok_idx[n] << 7) + (i & 127)];
}
__global__ void cat_kernel(const float* __restrict__ type_emb, const int* __restrict__ atype,
                           const float* __restrict__ shp, float* __restrict__ cat, int total) {
  int i = blockIdx.x * blockDim.x + threadIdx.x; if (i >= total) return;
  int a = i >> 7;
  cat[i] = type_emb[((size_t)atype[a] << 7) + (i & 127)] + shp[i];
}
__global__ void gate_kernel(float* __restrict__ agg, const float* __restrict__ g,
                            const float* __restrict__ s, int total) {
  int i = blockIdx.x * blockDim.x + threadIdx.x; if (i >= total) return;
  float a = agg[i];
  float gv = 1.0f / (1.0f + expf(-g[i]));
  agg[i] = a + gv * (s[i] - a);
}
__global__ void add_kernel(float* __restrict__ o, const float* __restrict__ a,
                           const float* __restrict__ b, int total) {
  int i = blockIdx.x * blockDim.x + threadIdx.x; if (i >= total) return;
  o[i] = a[i] + b[i];
}
__global__ void transpose_nodes_kernel(const float* __restrict__ in, float* __restrict__ out,
                                       int A_, int T_, int am2tm, int total) {
  int i = blockIdx.x * blockDim.x + threadIdx.x; if (i >= total) return;
  int n = i >> 7, d = i & 127;
  if (am2tm) { int a = n / T_, t = n % T_; out[((size_t)(t * A_ + a) << 7) + d] = in[i]; }
  else       { int t = n / A_, a = n % A_; out[((size_t)(a * T_ + t) << 7) + d] = in[i]; }
}

// ------------------------------ host helpers --------------------------------
static inline dim3 eg(int n) { return dim3((unsigned)((n + 255) / 256)); }

static inline void launch_gemm_f32(const float* A, int lda, const float* B, const float* bias,
                                   float* C, int N, int K, int NC, int flags, hipStream_t s) {
  dim3 grid(N / 64, NC / 32), blk(128);
  gemm_wmma_kernel<float><<<grid, blk, 0, s>>>(A, lda, B, bias, C, N, K, NC, flags);
}
static inline void launch_gemm_f16(const _Float16* A, int lda, const float* B, const float* bias,
                                   float* C, int N, int K, int NC, int flags, hipStream_t s) {
  dim3 grid(N / 64, NC / 32), blk(128);
  gemm_wmma_kernel<_Float16><<<grid, blk, 0, s>>>(A, lda, B, bias, C, N, K, NC, flags);
}
static inline void launch_ln(const float* x, const float* g, const float* b, float* y,
                             int N, int relu, hipStream_t s) {
  ln128_kernel<float, float><<<dim3((N + 7) / 8), 256, 0, s>>>(x, g, b, y, N, relu);
}
static inline void launch_ln16(const _Float16* x, const float* g, const float* b, _Float16* y,
                               int N, int relu, hipStream_t s) {
  ln128_kernel<_Float16, _Float16><<<dim3((N + 7) / 8), 256, 0, s>>>(x, g, b, y, N, relu);
}

// ------------------------------- entry point --------------------------------
extern "C" void kernel_launch(void* const* d_in, const int* in_sizes, int n_in,
                              void* d_out, int out_size, void* d_ws, size_t ws_size,
                              hipStream_t stream)
{
  (void)out_size;
  if (n_in < 251) return;

  const int A_   = in_sizes[0];
  const int NA   = in_sizes[2];           // A*T
  const int T_   = NA / A_;
  const int TOK_ = in_sizes[5] / 8;
  const int M_   = in_sizes[10];
  const int Et   = in_sizes[11] / 2;
  const int Ep   = in_sizes[12] / 2;
  const int Ea   = in_sizes[13] / 2;
  int maxE = Et > Ep ? Et : Ep; if (Ea > maxE) maxE = Ea;

  const int*   agent_type  = (const int*)  d_in[0];
  const float* agent_shape = (const float*)d_in[1];
  const int*   token_idx   = (const int*)  d_in[2];
  const float* token_pos   = (const float*)d_in[3];
  const float* token_head  = (const float*)d_in[4];
  const float* tok_raw[3]  = {(const float*)d_in[5], (const float*)d_in[6], (const float*)d_in[7]};
  const float* map_feat    = (const float*)d_in[8];
  const float* map_pos     = (const float*)d_in[9];
  const float* map_orient  = (const float*)d_in[10];
  const int* et  = (const int*)d_in[11];
  const int* epe = (const int*)d_in[12];
  const int* eae = (const int*)d_in[13];
  const int *src_t = et,  *dst_t = et  + Et;
  const int *src_p = epe, *dst_p = epe + Ep;
  const int *src_a = eae, *dst_a = eae + Ea;
  auto P = [&](int i) { return (const float*)d_in[i]; };

  // ---- workspace carve (bump allocator, 256B aligned) ----
  char* base = (char*)d_ws;
  size_t off = 0;
  auto alloc = [&](size_t bytes) -> void* {
    void* p = base + off; off += (bytes + 255) & ~(size_t)255; return p;
  };
  const size_t NA128 = (size_t)NA * 128;
  float* f_am = (float*)alloc(NA128 * 4);
  float* nb1  = (float*)alloc(NA128 * 4);
  float* nb2  = (float*)alloc(NA128 * 4);
  float* xs   = (float*)alloc(NA128 * 4);
  float* xd   = (float*)alloc(NA128 * 4);
  float* qb   = (float*)alloc(NA128 * 4);
  float* kb   = (float*)alloc(NA128 * 4);
  float* vb   = (float*)alloc(NA128 * 4);
  float* aggb = (float*)alloc(NA128 * 4);
  float* gbuf = (float*)alloc(NA128 * 4);
  float* sbuf = (float*)alloc(NA128 * 4);
  float* tmpb = (float*)alloc(NA128 * 4);
  float* xlnb = (float*)alloc(NA128 * 4);
  float* hff  = (float*)alloc((size_t)NA * 512 * 4);
  float* tokT[3];
  for (int i = 0; i < 3; ++i) tokT[i] = (float*)alloc((size_t)TOK_ * 128 * 4);
  float* tok_emb  = (float*)alloc(NA128 * 4);
  float* x_a      = (float*)alloc(NA128 * 4);
  float* catb     = (float*)alloc((size_t)A_ * 128 * 4);
  float* feats_xa = (float*)alloc((size_t)NA * 2 * 4);
  float* feats_rt = (float*)alloc((size_t)Et * 4 * 4);
  float* feats_rp = (float*)alloc((size_t)Ep * 3 * 4);
  float* feats_ra = (float*)alloc((size_t)Ea * 3 * 4);
  int maxNAtok = (TOK_ > A_) ? TOK_ : A_;
  float* Apad = (float*)alloc((size_t)maxNAtok * 32 * 4);       // col-padded small A
  float* Wpad = (float*)alloc((size_t)160 * 128 * 4);           // row-padded W1
  _Float16* r_t  = (_Float16*)alloc((size_t)Et * 128 * 2);
  _Float16* r_pt = (_Float16*)alloc((size_t)Ep * 128 * 2);
  _Float16* r_aa = (_Float16*)alloc((size_t)Ea * 128 * 2);
  _Float16* B1h  = (_Float16*)alloc((size_t)maxE * 160 * 2);   // fourier z / LN(r)
  float*    B2   = (float*)   alloc((size_t)maxE * 128 * 4);   // fourier h / kr
  float*    B3   = (float*)   alloc((size_t)maxE * 128 * 4);   // fourier hsum / vr
  if (off > ws_size) return;

  // ---- small mlp with K-padding: Linear(Ksrc->128) -> LN -> ReLU -> Linear ----
  auto mlp_small = [&](const float* X, int Ksrc, int N, int pb, float* outp) {
    pad_cols_kernel<<<eg(N * 32), 256, 0, stream>>>(X, Apad, Ksrc, 32, N * 32);
    pad_rows_kernel<<<eg(32 * 128), 256, 0, stream>>>(P(pb + 0), Wpad, Ksrc * 128, 32 * 128);
    launch_gemm_f32(Apad, 32, Wpad, P(pb + 1), tmpb, N, 32, 128, 0, stream);
    launch_ln(tmpb, P(pb + 2), P(pb + 3), tmpb, N, 1, stream);
    launch_gemm_f32(tmpb, 128, P(pb + 4), P(pb + 5), outp, N, 128, 128, 0, stream);
  };
  // ---- fourier embedding ----
  auto fourier = [&](const float* feats, int N, int C, int pb, float* outp, int outflags,
                     const float* cat) {
    const float* freqs = P(pb + 0);
    const float* W1 = P(pb + 1); const float* b1  = P(pb + 2);
    const float* g1 = P(pb + 3); const float* bt1 = P(pb + 4);
    const float* W2 = P(pb + 5); const float* b2  = P(pb + 6);
    const float* g2 = P(pb + 7); const float* bt2 = P(pb + 8);
    const float* W3 = P(pb + 9); const float* b3  = P(pb + 10);
    int total = N * 160;
    for (int ch = 0; ch < C; ++ch) {
      fourier_z_kernel<<<eg(total), 256, 0, stream>>>(feats, freqs, B1h, N, C, ch, total);
      pad_rows_kernel<<<eg(160 * 128), 256, 0, stream>>>(W1 + (size_t)ch * 129 * 128, Wpad,
                                                         129 * 128, 160 * 128);
      launch_gemm_f16(B1h, 160, Wpad, b1 + ch * 128, B2, N, 160, 128, 0, stream);
      launch_ln(B2, g1 + ch * 128, bt1 + ch * 128, B2, N, 1, stream);
      launch_gemm_f32(B2, 128, W2 + (size_t)ch * 128 * 128, b2 + ch * 128, B3, N, 128, 128,
                      ch == 0 ? 0 : FLAG_ACC, stream);
    }
    int tot128 = (int)((size_t)N * 128);
    if (cat) add_cat_kernel<<<eg(tot128), 256, 0, stream>>>(B3, cat, tot128, T_);
    launch_ln(B3, g2, bt2, B3, N, 1, stream);
    launch_gemm_f32(B3, 128, W3, b3, outp, N, 128, 128, outflags, stream);
  };
  // ---- graph attention layer ----
  auto attn = [&](int pb, const float* x_src, const float* x_dst, const _Float16* r,
                  const int* src, const int* dst, float* outp, int Ns, int Nd, int E) {
    const float *ln_s_g=P(pb+0), *ln_s_b=P(pb+1), *ln_d_g=P(pb+2), *ln_d_b=P(pb+3);
    const float *ln_r_g=P(pb+4), *ln_r_b=P(pb+5);
    const float *Wq=P(pb+6), *bq=P(pb+7), *Wk=P(pb+8), *Wv=P(pb+9), *bv=P(pb+10);
    const float *Wkr=P(pb+11), *Wvr=P(pb+12), *bvr=P(pb+13);
    const float *Wg=P(pb+14), *bg=P(pb+15), *Ws=P(pb+16), *bs=P(pb+17);
    const float *Wo=P(pb+18), *bo=P(pb+19), *ff_g=P(pb+20), *ff_b=P(pb+21);
    const float *W1=P(pb+22), *b1=P(pb+23), *W2=P(pb+24), *b2=P(pb+25);
    int nd128 = Nd * 128;
    launch_ln(x_src, ln_s_g, ln_s_b, xs, Ns, 0, stream);
    launch_ln(x_dst, ln_d_g, ln_d_b, xd, Nd, 0, stream);
    launch_ln16(r, ln_r_g, ln_r_b, B1h, E, 0, stream);
    launch_gemm_f32(xd, 128, Wq, bq, qb, Nd, 128, 128, 0, stream);
    launch_gemm_f32(xs, 128, Wk, nullptr, kb, Ns, 128, 128, 0, stream);
    launch_gemm_f32(xs, 128, Wv, bv, vb, Ns, 128, 128, 0, stream);
    launch_gemm_f16(B1h, 128, Wkr, nullptr, B2, E, 128, 128, 0, stream);
    launch_gemm_f16(B1h, 128, Wvr, bvr, B3, E, 128, 128, 0, stream);
    attn_gather_kernel<<<dim3((Nd + 7) / 8), 256, 0, stream>>>(qb, kb, vb, B2, B3,
                                                               src, dst, aggb, E, Nd);
    launch_gemm_f32(aggb, 128, Wg, bg, gbuf, Nd, 128, 128, 0, stream);
    launch_gemm_f32(xd, 128, Wg + 128 * 128, nullptr, gbuf, Nd, 128, 128, FLAG_ACC, stream);
    launch_gemm_f32(xd, 128, Ws, bs, sbuf, Nd, 128, 128, 0, stream);
    gate_kernel<<<eg(nd128), 256, 0, stream>>>(aggb, gbuf, sbuf, nd128);
    launch_gemm_f32(aggb, 128, Wo, bo, tmpb, Nd, 128, 128, 0, stream);
    add_kernel<<<eg(nd128), 256, 0, stream>>>(outp, x_dst, tmpb, nd128);
    launch_ln(outp, ff_g, ff_b, xlnb, Nd, 0, stream);
    launch_gemm_f32(xlnb, 128, W1, b1, hff, Nd, 128, 512, FLAG_RELU, stream);
    launch_gemm_f32(hff, 512, W2, b2, outp, Nd, 512, 128, FLAG_ACC, stream);
  };

  // ================= forward =================
  const int na128 = (int)NA128;
  // token category embedding tables (params: tok_veh=65, tok_ped=71, tok_cyc=77)
  for (int i = 0; i < 3; ++i) mlp_small(tok_raw[i], 8, TOK_, 65 + 6 * i, tokT[i]);
  tok_gather_kernel<<<eg(na128), 256, 0, stream>>>(tokT[0], tokT[1], tokT[2],
                                                   token_idx, agent_type, tok_emb, na128, T_);
  // motion features + agent category embedding
  feat_xa_kernel<<<eg(NA), 256, 0, stream>>>(token_pos, token_head, feats_xa, NA, T_);
  mlp_small(agent_shape, 3, A_, 15, qb);                         // shape_emb -> qb
  cat_kernel<<<eg(A_ * 128), 256, 0, stream>>>(P(14), agent_type, qb, catb, A_ * 128);
  fourier(feats_xa, NA, 2, 21, x_a, 0, catb);                    // x_a_emb
  // fusion mlp over concat(tok_emb, x_a)
  launch_gemm_f32(tok_emb, 128, P(83), P(84), tmpb, NA, 128, 128, 0, stream);
  launch_gemm_f32(x_a, 128, P(83) + 128 * 128, nullptr, tmpb, NA, 128, 128, FLAG_ACC, stream);
  launch_ln(tmpb, P(85), P(86), tmpb, NA, 1, stream);
  launch_gemm_f32(tmpb, 128, P(87), P(88), f_am, NA, 128, 128, 0, stream);
  // relational embeddings (stored f16)
  feat_rt_kernel<<<eg(Et), 256, 0, stream>>>(token_pos, token_head, src_t, dst_t, feats_rt, Et, T_);
  feat_pt_kernel<<<eg(Ep), 256, 0, stream>>>(map_pos, map_orient, token_pos, token_head,
                                             src_p, dst_p, feats_rp, Ep, A_, T_);
  feat_aa_kernel<<<eg(Ea), 256, 0, stream>>>(token_pos, token_head, src_a, dst_a, feats_ra, Ea, A_, T_);
  fourier(feats_rt, Et, 4, 32, (float*)r_t,  FLAG_F16OUT, nullptr);
  fourier(feats_rp, Ep, 3, 43, (float*)r_pt, FLAG_F16OUT, nullptr);
  fourier(feats_ra, Ea, 3, 54, (float*)r_aa, FLAG_F16OUT, nullptr);
  // L=2 rounds of temporal / map->agent / agent->agent attention
  for (int i = 0; i < 2; ++i) {
    attn(95  + 26 * i, f_am, f_am, r_t, src_t, dst_t, nb1, NA, NA, Et);
    transpose_nodes_kernel<<<eg(na128), 256, 0, stream>>>(nb1, nb2, A_, T_, 1, na128);
    attn(147 + 26 * i, map_feat, nb2, r_pt, src_p, dst_p, nb1, M_, NA, Ep);
    attn(199 + 26 * i, nb1, nb1, r_aa, src_a, dst_a, nb2, NA, NA, Ea);
    transpose_nodes_kernel<<<eg(na128), 256, 0, stream>>>(nb2, f_am, A_, T_, 0, na128);
  }
  // prediction head -> [A*T, TOK] logits
  launch_gemm_f32(f_am, 128, P(89), P(90), tmpb, NA, 128, 128, 0, stream);
  launch_ln(tmpb, P(91), P(92), tmpb, NA, 1, stream);
  launch_gemm_f32(tmpb, 128, P(93), P(94), (float*)d_out, NA, 128, TOK_, 0, stream);
}
